// Block_Head_34832184771061
// MI455X (gfx1250) — compile-verified, matrix operand order checked by reading, not run on hardware
//
#include <hip/hip_runtime.h>
#include <math.h>

#define T_SEQ   384
#define DMODEL  768
#define DINNER  3072
#define DSTATE  128
#define DTRANK  48
#define DTPAD   64
#define XDBL    304          // logical cols of x_dbl
#define XDBLP   384          // padded row stride (3x128) for tiled WMMA GEMM
#define HMLP    768

typedef __attribute__((ext_vector_type(16))) _Float16 v16h;
typedef __attribute__((ext_vector_type(8)))  _Float16 v8h;
typedef __attribute__((ext_vector_type(4)))  _Float16 v4h;
typedef __attribute__((ext_vector_type(8)))  float    v8f;

__device__ __forceinline__ float siluf(float x) { return x / (1.f + __expf(-x)); }

// ---------------------------------------------------------------------------
// Tiled GEMM:  C[M x N] = A[M x K] * W[N x K]^T  (fp32 in, f16 WMMA, f32 acc)
// Workgroup: 256 threads (8 waves). Output tile 128(M) x 128(N), K-step 32.
// LDS staging in f16 (padded rows: no bank conflicts on 64-bank LDS); each
// wave computes 32x64 via 8 WMMAs (2 A-frags x 4 B-frags, full register
// reuse). grid = (N/128, M/128, batch). K%32==0, M%128==0, N%128==0.
// ---------------------------------------------------------------------------
#define MT   128
#define NT   128
#define LDH  40              // 32 halves + 8 pad (80B row stride)

__global__ void __launch_bounds__(256)
gemm_tn_wmma_tiled(const float* __restrict__ A, const float* __restrict__ W,
                   float* __restrict__ C, int K, int N,
                   long long sA, long long sW, long long sC)
{
    __shared__ _Float16 sAh[MT * LDH];
    __shared__ _Float16 sBh[NT * LDH];

    const int tid = threadIdx.x;
    const int n0  = blockIdx.x * NT;
    const int m0  = blockIdx.y * MT;
    const int b   = blockIdx.z;
    const float* gA = A + b * sA;
    const float* gW = W + b * sW;

    const int lane = tid & 31, wave = tid >> 5;
    const int wm = wave & 3, wn = wave >> 2;       // 4 M-groups(32) x 2 N-groups(64)
    const int r  = lane & 15, hi = lane >> 4;

    float4 ra[4], rb[4];                           // 128x32 f32 = 1024 float4 each

    auto loadA = [&](int k0) {
#pragma unroll
        for (int j = 0; j < 4; ++j) {
            int q = tid + j * 256;
            int row = q >> 3, c4 = q & 7;
            ra[j] = *(const float4*)(gA + (long long)(m0 + row) * K + k0 + c4 * 4);
        }
    };
    auto loadB = [&](int k0) {
#pragma unroll
        for (int j = 0; j < 4; ++j) {
            int q = tid + j * 256;
            int row = q >> 3, c4 = q & 7;
            rb[j] = *(const float4*)(gW + (long long)(n0 + row) * K + k0 + c4 * 4);
        }
    };
    auto stage = [&]() {
#pragma unroll
        for (int j = 0; j < 4; ++j) {
            int q = tid + j * 256;
            int row = q >> 3, c4 = q & 7;
            v4h ha, hb;
            ha[0] = (_Float16)ra[j].x; ha[1] = (_Float16)ra[j].y;
            ha[2] = (_Float16)ra[j].z; ha[3] = (_Float16)ra[j].w;
            hb[0] = (_Float16)rb[j].x; hb[1] = (_Float16)rb[j].y;
            hb[2] = (_Float16)rb[j].z; hb[3] = (_Float16)rb[j].w;
            *(v4h*)&sAh[row * LDH + c4 * 4] = ha;
            *(v4h*)&sBh[row * LDH + c4 * 4] = hb;
        }
    };
    auto fragload = [&](const _Float16* base, int row) -> v16h {
        v8h lo = *(const v8h*)(base + row * LDH + hi * 8);
        v8h hh = *(const v8h*)(base + row * LDH + hi * 8 + 16);
        return __builtin_shufflevector(lo, hh, 0, 1, 2, 3, 4, 5, 6, 7,
                                               8, 9, 10, 11, 12, 13, 14, 15);
    };

    v8f acc[2][4];
#pragma unroll
    for (int i = 0; i < 2; ++i)
#pragma unroll
        for (int nb = 0; nb < 4; ++nb) acc[i][nb] = (v8f){};

    loadA(0);
    loadB(0);
    for (int k0 = 0; k0 < K; k0 += 32) {
        __syncthreads();
        stage();
        __syncthreads();
        if (k0 + 32 < K) {                         // overlap HBM with WMMA
            __builtin_prefetch(gA + (long long)(m0 + (tid >> 3)) * K + k0 + 64, 0, 0);
            __builtin_prefetch(gW + (long long)(n0 + (tid >> 3)) * K + k0 + 64, 0, 0);
            loadA(k0 + 32);
            loadB(k0 + 32);
        }
        v16h af0 = fragload(sAh, wm * 32 + r);
        v16h af1 = fragload(sAh, wm * 32 + 16 + r);
#pragma unroll
        for (int nb = 0; nb < 4; ++nb) {
            v16h bf = fragload(sBh, wn * 64 + nb * 16 + r);
            acc[0][nb] = __builtin_amdgcn_wmma_f32_16x16x32_f16(
                             false, af0, false, bf, (short)0, acc[0][nb], false, false);
            acc[1][nb] = __builtin_amdgcn_wmma_f32_16x16x32_f16(
                             false, af1, false, bf, (short)0, acc[1][nb], false, false);
        }
    }

    float* pc = C + b * sC + (long long)(m0 + wm * 32 + hi * 8) * N + n0 + wn * 64 + r;
#pragma unroll
    for (int i = 0; i < 2; ++i)
#pragma unroll
        for (int nb = 0; nb < 4; ++nb)
#pragma unroll
            for (int rr = 0; rr < 8; ++rr)
                pc[(long long)(i * 16 + rr) * N + nb * 16] = acc[i][nb][rr];
}

// ---------------------------------------------------------------------------
// RMSNorm over 768 cols. One 256-thread block per row. grid=(rows,1,3).
// ---------------------------------------------------------------------------
__global__ void __launch_bounds__(256)
rmsnorm_k(const float* __restrict__ in, const float* __restrict__ lnw,
          float* __restrict__ out, long long sIn, long long sOut, int which)
{
    __shared__ float red[256];
    const int row = blockIdx.x, b = blockIdx.z, tid = threadIdx.x;
    const float* pin = in + b * sIn + (long long)row * DMODEL;
    const float* w   = lnw + (2 * b + which) * DMODEL;
    float x0 = pin[tid], x1 = pin[tid + 256], x2 = pin[tid + 512];
    red[tid] = x0 * x0 + x1 * x1 + x2 * x2;
    __syncthreads();
    for (int s = 128; s > 0; s >>= 1) {
        if (tid < s) red[tid] += red[tid + s];
        __syncthreads();
    }
    float scale = rsqrtf(red[0] * (1.f / 768.f) + 1e-6f);
    float* po = out + b * sOut + (long long)row * DMODEL;
    po[tid]       = x0 * scale * w[tid];
    po[tid + 256] = x1 * scale * w[tid + 256];
    po[tid + 512] = x2 * scale * w[tid + 512];
}

// ---------------------------------------------------------------------------
// Causal depthwise conv (width 4) + bias + SiLU on x half of xz.
// ---------------------------------------------------------------------------
__global__ void __launch_bounds__(256)
conv_silu_k(const float* __restrict__ xz, const float* __restrict__ cw,
            const float* __restrict__ cb, float* __restrict__ xc)
{
    int idx = blockIdx.x * 256 + threadIdx.x;      // t*DINNER + d
    int b = blockIdx.y;
    int t = idx / DINNER, d = idx - t * DINNER;
    const float* px = xz + (long long)b * T_SEQ * (2 * DINNER);
    const float* w  = cw + ((long long)b * DINNER + d) * 4;
    float acc = cb[b * DINNER + d];
#pragma unroll
    for (int j = 0; j < 4; ++j) {
        int tt = t - 3 + j;
        if (tt >= 0) acc += w[j] * px[(long long)tt * (2 * DINNER) + d];
    }
    xc[(long long)b * T_SEQ * DINNER + idx] = siluf(acc);
}

// Copy rows with zero-padded K (dt 48->64 from padded x_dbl, dt_proj_w 48->64)
__global__ void __launch_bounds__(256)
padcopy_k(const float* __restrict__ src, float* __restrict__ dst,
          int srcStride, int sk, int dk, int n, long long sSrc, long long sDst)
{
    int idx = blockIdx.x * 256 + threadIdx.x;
    if (idx >= n) return;
    int b = blockIdx.y;
    int r = idx / dk, j = idx - r * dk;
    float v = (j < sk) ? src[b * sSrc + (long long)r * srcStride + j] : 0.f;
    dst[b * sDst + idx] = v;
}

// Zero-pad rows: dst[dstRows x cols], rows >= srcRows are zero (x_proj_w 304->384)
__global__ void __launch_bounds__(256)
padrows_k(const float* __restrict__ src, float* __restrict__ dst,
          int srcRows, int cols, int dstRows)
{
    int idx = blockIdx.x * 256 + threadIdx.x;      // r*cols + c
    int b = blockIdx.y;
    int r = idx / cols, c = idx - r * cols;
    float v = (r < srcRows) ? src[(long long)b * srcRows * cols + (long long)r * cols + c] : 0.f;
    dst[(long long)b * dstRows * cols + idx] = v;
}

__global__ void __launch_bounds__(256)
softplus_bias_k(float* __restrict__ delta, const float* __restrict__ dtb)
{
    int idx = blockIdx.x * 256 + threadIdx.x;      // t*DINNER + d
    int b = blockIdx.y;
    int d = idx % DINNER;
    long long p = (long long)b * T_SEQ * DINNER + idx;
    float x = delta[p] + dtb[b * DINNER + d];
    delta[p] = (x > 20.f) ? x : log1pf(__expf(x));
}

// ---------------------------------------------------------------------------
// Selective scan: one wave per channel d, 4 states per lane, sequential in T.
// x_dbl rows have stride XDBLP; B at col 48, C at col 176 (16B aligned).
// ---------------------------------------------------------------------------
__global__ void __launch_bounds__(32)
scan_k(const float* __restrict__ delta, const float* __restrict__ xdbl,
       const float* __restrict__ u, const float* __restrict__ A_log,
       const float* __restrict__ D_skip, float* __restrict__ y)
{
    const int d = blockIdx.x, b = blockIdx.z, lane = threadIdx.x;
    const float* dl = delta + (long long)b * T_SEQ * DINNER;
    const float* xd = xdbl  + (long long)b * T_SEQ * XDBLP;
    const float* uu = u     + (long long)b * T_SEQ * DINNER;
    float*       yy = y     + (long long)b * T_SEQ * DINNER;
    const float* Arow = A_log + ((long long)b * DINNER + d) * DSTATE;
    const float  Dv = D_skip[b * DINNER + d];

    float Ac[4], h[4] = {0.f, 0.f, 0.f, 0.f};
#pragma unroll
    for (int j = 0; j < 4; ++j) Ac[j] = -__expf(Arow[lane * 4 + j]);

    for (int t = 0; t < T_SEQ; ++t) {
        float dt = dl[(long long)t * DINNER + d];
        float ut = uu[(long long)t * DINNER + d];
        const float4 Bv = *(const float4*)(xd + (long long)t * XDBLP + DTRANK          + lane * 4);
        const float4 Cv = *(const float4*)(xd + (long long)t * XDBLP + DTRANK + DSTATE + lane * 4);
        float du = dt * ut, yp;
        h[0] = __expf(dt * Ac[0]) * h[0] + du * Bv.x;  yp  = h[0] * Cv.x;
        h[1] = __expf(dt * Ac[1]) * h[1] + du * Bv.y;  yp += h[1] * Cv.y;
        h[2] = __expf(dt * Ac[2]) * h[2] + du * Bv.z;  yp += h[2] * Cv.z;
        h[3] = __expf(dt * Ac[3]) * h[3] + du * Bv.w;  yp += h[3] * Cv.w;
#pragma unroll
        for (int off = 16; off >= 1; off >>= 1) yp += __shfl_xor(yp, off, 32);
        if (lane == 0) yy[(long long)t * DINNER + d] = yp + ut * Dv;
    }
}

__global__ void __launch_bounds__(256)
gate_k(const float* __restrict__ y, const float* __restrict__ xz, float* __restrict__ yg)
{
    int idx = blockIdx.x * 256 + threadIdx.x;      // t*DINNER + d
    int b = blockIdx.y;
    int t = idx / DINNER, d = idx - t * DINNER;
    float z = xz[(long long)b * T_SEQ * (2 * DINNER) + (long long)t * (2 * DINNER) + DINNER + d];
    long long p = (long long)b * T_SEQ * DINNER + idx;
    yg[p] = y[p] * siluf(z);
}

__global__ void __launch_bounds__(256)
add2_k(const float* __restrict__ a, const float* __restrict__ c, float* __restrict__ o, long long s)
{
    int idx = blockIdx.x * 256 + threadIdx.x;
    long long p = (long long)blockIdx.y * s + idx;
    o[p] = a[p] + c[p];
}

__global__ void __launch_bounds__(256)
glu_k(const float* __restrict__ h1, const float* __restrict__ b1, float* __restrict__ act)
{
    int idx = blockIdx.x * 256 + threadIdx.x;      // t*HMLP + j
    int b = blockIdx.y;
    int t = idx / HMLP, j = idx - t * HMLP;
    const float* ph = h1 + (long long)b * T_SEQ * (2 * HMLP) + (long long)t * (2 * HMLP);
    float a = ph[j]        + b1[b * 2 * HMLP + j];
    float g = ph[HMLP + j] + b1[b * 2 * HMLP + HMLP + j];
    act[(long long)b * T_SEQ * HMLP + idx] = a * siluf(g);
}

__global__ void __launch_bounds__(256)
final_k(const float* __restrict__ bres, const float* __restrict__ fc2o,
        const float* __restrict__ b2, float* __restrict__ out)
{
    int idx = blockIdx.x * 256 + threadIdx.x;      // t*DMODEL + c
    int b = blockIdx.y;
    int c = idx % DMODEL;
    long long p = (long long)b * T_SEQ * DMODEL + idx;
    out[p] = bres[p] + fc2o[p] + b2[b * DMODEL + c];
}

// ---------------------------------------------------------------------------
extern "C" void kernel_launch(void* const* d_in, const int* in_sizes, int n_in,
                              void* d_out, int out_size, void* d_ws, size_t ws_size,
                              hipStream_t stream)
{
    const float* x    = (const float*)d_in[0];
    const float* lnw  = (const float*)d_in[1];
    const float* inw  = (const float*)d_in[2];
    const float* cw   = (const float*)d_in[3];
    const float* cb   = (const float*)d_in[4];
    const float* xpw  = (const float*)d_in[5];
    const float* dtw  = (const float*)d_in[6];
    const float* dtb  = (const float*)d_in[7];
    const float* Alog = (const float*)d_in[8];
    const float* Dsk  = (const float*)d_in[9];
    const float* outw = (const float*)d_in[10];
    const float* w1   = (const float*)d_in[11];
    const float* b1   = (const float*)d_in[12];
    const float* w2   = (const float*)d_in[13];
    const float* b2   = (const float*)d_in[14];
    float* out = (float*)d_out;

    float* ws = (float*)d_ws;
    size_t o = 0;
    float* xn    = ws + o; o += 3ull * T_SEQ * DMODEL;
    float* xz    = ws + o; o += 3ull * T_SEQ * 2 * DINNER;
    float* xconv = ws + o; o += 3ull * T_SEQ * DINNER;
    float* xdbl  = ws + o; o += 3ull * T_SEQ * XDBLP;      // padded stride
    float* xpwp  = ws + o; o += 3ull * XDBLP * DINNER;     // padded x_proj_w
    float* dtp   = ws + o; o += 3ull * T_SEQ * DTPAD;
    float* dtwp  = ws + o; o += 3ull * DINNER * DTPAD;
    float* delta = ws + o; o += 3ull * T_SEQ * DINNER;
    float* yb    = ws + o; o += 3ull * T_SEQ * DINNER;
    float* yg    = ws + o; o += 3ull * T_SEQ * DINNER;
    float* mo    = ws + o; o += 3ull * T_SEQ * DMODEL;
    float* bres  = ws + o; o += 3ull * T_SEQ * DMODEL;
    float* bn    = ws + o; o += 3ull * T_SEQ * DMODEL;
    float* h1    = ws + o; o += 3ull * T_SEQ * 2 * HMLP;
    float* act   = ws + o; o += 3ull * T_SEQ * HMLP;
    float* fc2o  = ws + o; o += 3ull * T_SEQ * DMODEL;

    const long long sTM = (long long)T_SEQ * DMODEL;
    const long long sXZ = (long long)T_SEQ * 2 * DINNER;
    const long long sTI = (long long)T_SEQ * DINNER;
    const dim3 blk(256);
    const dim3 gTI((T_SEQ * DINNER) / 256, 3);
    const dim3 gTM((T_SEQ * DMODEL) / 256, 3);

    // 1) pre-mamba RMSNorm (chunks are batch z)
    rmsnorm_k<<<dim3(T_SEQ, 1, 3), blk, 0, stream>>>(x, lnw, xn, sTM, sTM, 0);

    // 2) in_proj: (384x768)(768x6144^T) -> xz
    gemm_tn_wmma_tiled<<<dim3(2 * DINNER / NT, T_SEQ / MT, 3), blk, 0, stream>>>(
        xn, inw, xz, DMODEL, 2 * DINNER, sTM, (long long)2 * DINNER * DMODEL, sXZ);

    // 3) causal depthwise conv + SiLU
    conv_silu_k<<<gTI, blk, 0, stream>>>(xz, cw, cb, xconv);

    // 4) x_proj (N padded 304->384): pad weight rows once, then tiled WMMA
    padrows_k<<<dim3((XDBLP * DINNER) / 256, 3), blk, 0, stream>>>(
        xpw, xpwp, XDBL, DINNER, XDBLP);
    gemm_tn_wmma_tiled<<<dim3(XDBLP / NT, T_SEQ / MT, 3), blk, 0, stream>>>(
        xconv, xpwp, xdbl, DINNER, XDBLP, sTI,
        (long long)XDBLP * DINNER, (long long)T_SEQ * XDBLP);

    // 5) zero-pad dt (K 48->64) and dt_proj_w
    padcopy_k<<<dim3((T_SEQ * DTPAD) / 256, 3), blk, 0, stream>>>(
        xdbl, dtp, XDBLP, DTRANK, DTPAD, T_SEQ * DTPAD,
        (long long)T_SEQ * XDBLP, (long long)T_SEQ * DTPAD);
    padcopy_k<<<dim3((DINNER * DTPAD) / 256, 3), blk, 0, stream>>>(
        dtw, dtwp, DTRANK, DTRANK, DTPAD, DINNER * DTPAD,
        (long long)DINNER * DTRANK, (long long)DINNER * DTPAD);

    // 6) dt_proj: (384x64)(3072x64^T) -> delta ; softplus(+bias)
    gemm_tn_wmma_tiled<<<dim3(DINNER / NT, T_SEQ / MT, 3), blk, 0, stream>>>(
        dtp, dtwp, delta, DTPAD, DINNER,
        (long long)T_SEQ * DTPAD, (long long)DINNER * DTPAD, sTI);
    softplus_bias_k<<<gTI, blk, 0, stream>>>(delta, dtb);

    // 7) selective scan (one wave32 per channel, batched over 3 blocks)
    scan_k<<<dim3(DINNER, 1, 3), dim3(32), 0, stream>>>(delta, xdbl, xconv, Alog, Dsk, yb);

    // 8) gate with silu(z), out_proj, residual
    gate_k<<<gTI, blk, 0, stream>>>(yb, xz, yg);
    gemm_tn_wmma_tiled<<<dim3(DMODEL / NT, T_SEQ / MT, 3), blk, 0, stream>>>(
        yg, outw, mo, DINNER, DMODEL, sTI, (long long)DMODEL * DINNER, sTM);
    add2_k<<<gTM, blk, 0, stream>>>(x, mo, bres, sTM);

    // 9) gated MLP
    rmsnorm_k<<<dim3(T_SEQ, 1, 3), blk, 0, stream>>>(bres, lnw, bn, sTM, sTM, 1);
    gemm_tn_wmma_tiled<<<dim3(2 * HMLP / NT, T_SEQ / MT, 3), blk, 0, stream>>>(
        bn, w1, h1, DMODEL, 2 * HMLP, sTM, (long long)2 * HMLP * DMODEL,
        (long long)T_SEQ * 2 * HMLP);
    glu_k<<<dim3((T_SEQ * HMLP) / 256, 3), blk, 0, stream>>>(h1, b1, act);
    gemm_tn_wmma_tiled<<<dim3(DMODEL / NT, T_SEQ / MT, 3), blk, 0, stream>>>(
        act, w2, fc2o, HMLP, DMODEL, (long long)T_SEQ * HMLP,
        (long long)DMODEL * HMLP, sTM);

    // 10) final residual + fc2 bias -> d_out (chunks concatenated)
    final_k<<<gTM, blk, 0, stream>>>(bres, fc2o, b2, out);
}
